// skipgram_23708219474347
// MI455X (gfx1250) — matrix operand order repaired and checked by reading, not compiled, not run on hardware
//
#include <hip/hip_runtime.h>
#include <hip/hip_bf16.h>

#define DIM   128
#define BATCH 16384
#define NEG   20
#define REGC  1e-6
#define TILE  16
#define TPC   7     // targets per wave chunk (3 chunks * 7 = 21 targets)

typedef __attribute__((ext_vector_type(16))) __bf16       v16bf;
typedef __attribute__((ext_vector_type(8)))  __bf16       v8bf;
typedef __attribute__((ext_vector_type(8)))  float        v8f;
typedef __attribute__((ext_vector_type(8)))  float        v8fv;
typedef __attribute__((ext_vector_type(16))) float        v16f;

// Convert 16 f32 (two 32B-aligned runs of 8) -> v16bf using HW bf16 converters
__device__ __forceinline__ v16bf cvt_2x8(const float* p0, const float* p1) {
    v8bf lo = __builtin_convertvector(*(const v8fv*)p0, v8bf);
    v8bf hi = __builtin_convertvector(*(const v8fv*)p1, v8bf);
    return __builtin_shufflevector(lo, hi, 0, 1, 2, 3, 4, 5, 6, 7,
                                   8, 9, 10, 11, 12, 13, 14, 15);
}
__device__ __forceinline__ float logsig(float x) {
    // numerically stable log(sigmoid(x))
    return fminf(x, 0.0f) - log1pf(__expf(-fabsf(x)));
}

__global__ void sg_init_kernel(double* acc) {
    if (threadIdx.x < 2) acc[threadIdx.x] = 0.0;
}

// REG term: sum |u_emb[u_pos[b]][d] - pretrained[u_pos[b]][d]| over all b,d
__global__ __launch_bounds__(256) void sg_reg_kernel(
    const float* __restrict__ u_emb, const float* __restrict__ pre,
    const int* __restrict__ u_pos, double* __restrict__ acc)
{
    const int N = BATCH * DIM;
    float part = 0.0f;
    for (int i = blockIdx.x * blockDim.x + threadIdx.x; i < N;
         i += gridDim.x * blockDim.x) {
        int b = i >> 7;
        int d = i & (DIM - 1);
        long r = (long)u_pos[b] * DIM + d;
        part += fabsf(u_emb[r] - pre[r]);
    }
    for (int off = 16; off; off >>= 1) part += __shfl_xor(part, off);
    if ((threadIdx.x & 31) == 0) atomicAdd(acc + 1, (double)part);
}

// Score kernel: each wave = (tile of 16 batch elems, chunk of 7 targets).
// Diagonal of D = A(u rows) x B(per-element target columns) gives the 16 dots.
__global__ __launch_bounds__(256) void sg_score_kernel(
    const float* __restrict__ u_emb, const float* __restrict__ v_emb,
    const int* __restrict__ u_pos, const int* __restrict__ v_pos,
    const int* __restrict__ v_neg, double* __restrict__ acc)
{
    const int lane = threadIdx.x & 31;
    const int wave = blockIdx.x * (blockDim.x >> 5) + (threadIdx.x >> 5);
    const int tile  = wave / 3;
    const int chunk = wave % 3;
    const int hl = lane >> 4;      // half-wave selector
    const int m  = lane & 15;      // A row / B column owned by this lane

    // ---- Build A (16x32 bf16 per K-chunk) for the 16 u-rows of this tile ----
    // Lane layout: elems 0..15 -> K = kc*32 + hl*16 + i
    const long rowA = (long)u_pos[tile * TILE + m] * DIM;
    v16bf amat[4];
#pragma unroll
    for (int kc = 0; kc < 4; ++kc) {
        const float* p0 = u_emb + rowA + kc * 32 + hl * 16;
        amat[kc] = cvt_2x8(p0, p0 + 8);
    }

    // ---- Precompute the 7 target row indices for our column ----
    const int bB = tile * TILE + m;    // batch element for our B column
    int tidx[TPC];
#pragma unroll
    for (int jj = 0; jj < TPC; ++jj) {
        const int j = chunk * TPC + jj;            // 0 = positive, 1..20 = negatives
        tidx[jj] = (j == 0) ? v_pos[bB] : v_neg[bB * NEG + (j - 1)];
    }

    float wpart = 0.0f;
#pragma unroll
    for (int jj = 0; jj < TPC; ++jj) {
        const int j = chunk * TPC + jj;
        // B layout: lane holds 16 consecutive K starting at kc*32 + hl*16, column m
        const float* w = v_emb + (long)tidx[jj] * DIM + hl * 16;

        // 1) issue all 16 b128 loads for this target's row slice
        v16f raw[4];
#pragma unroll
        for (int kc = 0; kc < 4; ++kc) raw[kc] = *(const v16f*)(w + kc * 32);
        // 2) convert all 4 chunks into distinct register sets
        v16bf bm[4];
#pragma unroll
        for (int kc = 0; kc < 4; ++kc)
            bm[kc] = __builtin_convertvector(raw[kc], v16bf);
        // 3) back-to-back accumulating WMMAs (no D->A/B hazard in the chain)
        v8f c = {};
#pragma unroll
        for (int kc = 0; kc < 4; ++kc)
            c = __builtin_amdgcn_wmma_f32_16x16x32_bf16(
                    false, amat[kc], false, bm[kc], (short)0, c, false, false);

        // Diagonal D[n][n]: lanes 0..7 hold it at c[lane] (n=lane);
        // lanes 24..31 hold it at c[lane-24] (n=lane-16).
        float arr[8];
#pragma unroll
        for (int i = 0; i < 8; ++i) arr[i] = c[i];
        const bool active = (lane < 8) || (lane >= 24);
        const int  di = (lane < 8) ? lane : ((lane >= 24) ? (lane - 24) : 0);
        const float s = arr[di];
        const float sgn = (j == 0) ? 1.0f : -1.0f;
        wpart += active ? logsig(sgn * s) : 0.0f;
    }

    for (int off = 16; off; off >>= 1) wpart += __shfl_xor(wpart, off);
    if (lane == 0) atomicAdd(acc, (double)wpart);
}

__global__ void sg_final_kernel(const double* acc, const int* bs, float* out) {
    if (threadIdx.x == 0) {
        double B = (double)bs[0];
        out[0] = (float)(-(acc[0] / B) - REGC * acc[1]);
    }
}

extern "C" void kernel_launch(void* const* d_in, const int* in_sizes, int n_in,
                              void* d_out, int out_size, void* d_ws, size_t ws_size,
                              hipStream_t stream) {
    const float* u_emb = (const float*)d_in[0];
    const float* v_emb = (const float*)d_in[1];
    const float* pre   = (const float*)d_in[2];
    const int*   u_pos = (const int*)d_in[3];
    const int*   v_pos = (const int*)d_in[4];
    const int*   v_neg = (const int*)d_in[5];
    const int*   bsz   = (const int*)d_in[6];
    float*  out = (float*)d_out;
    double* acc = (double*)d_ws;   // acc[0]=main log-sum, acc[1]=reg L1 sum

    sg_init_kernel<<<1, 32, 0, stream>>>(acc);
    sg_reg_kernel<<<1024, 256, 0, stream>>>(u_emb, pre, u_pos, acc);
    // (BATCH/TILE)=1024 tiles * 3 chunks = 3072 waves; 8 waves/block -> 384 blocks
    sg_score_kernel<<<384, 256, 0, stream>>>(u_emb, v_emb, u_pos, v_pos, v_neg, acc);
    sg_final_kernel<<<1, 32, 0, stream>>>(acc, bsz, out);
}
